// GroupedQueryAttention_46574625358207
// MI455X (gfx1250) — compile-verified, matrix operand order checked by reading, not run on hardware
//
#include <hip/hip_runtime.h>
#include <hip/hip_bf16.h>
#include <stddef.h>

// ---------------------------------------------------------------------------
// GQA attention for MI455X (gfx1250, wave32, WMMA f32_16x16x32_f16,
// async global->LDS staging). Sizes fixed by the reference: B=8, T=16,
// D=4096, H=32, G=8, R=4, Dh=128, MAX_SEQ=8192. start_pos read on device.
// ---------------------------------------------------------------------------

typedef __attribute__((ext_vector_type(16))) _Float16 v16h;
typedef __attribute__((ext_vector_type(8)))  float    v8f;
typedef int v4i_t __attribute__((vector_size(4 * sizeof(int))));

#define NSPLIT 8            // S-splits in the attention kernel
#define SCHUNK (8192 / NSPLIT)
#define KV_PITCH 132        // padded LDS row pitch (floats); 528B, 16B-aligned

__device__ __forceinline__ v8f wmma_f16(v16h a, v16h b, v8f c) {
  // D = A(16x32,f16) * B(32x16,f16) + C(16x16,f32)
  return __builtin_amdgcn_wmma_f32_16x16x32_f16(
      /*neg_a=*/false, a, /*neg_b=*/false, b,
      /*c_mod=*/(short)0, c, /*reuse_a=*/false, /*reuse_b=*/false);
}

// Async 16-byte global -> LDS copy (per-lane addresses), ASYNCcnt-tracked.
// Builtin signature (from hipcc diagnostic): param0 = addrspace(1) v4i*
// (global source), so ordering is (global src, lds dst, imm offset, imm cpol).
__device__ __forceinline__ void async_cp16(const float* g, float* l) {
#if __has_builtin(__builtin_amdgcn_global_load_async_to_lds_b128)
  __builtin_amdgcn_global_load_async_to_lds_b128(
      (__attribute__((address_space(1))) v4i_t*)const_cast<float*>(g),
      (__attribute__((address_space(3))) v4i_t*)l,
      /*offset=*/0, /*cpol=*/0);
#else
  unsigned loff =
      (unsigned)(size_t)(__attribute__((address_space(3))) void*)l;
  asm volatile("global_load_async_to_lds_b128 %0, %1, off"
               :: "v"(loff), "v"(g) : "memory");
#endif
}

__device__ __forceinline__ void wait_async0() {
#if __has_builtin(__builtin_amdgcn_s_wait_asynccnt)
  __builtin_amdgcn_s_wait_asynccnt(0);
#else
  asm volatile("s_wait_asynccnt 0x0" ::: "memory");
#endif
}

// Build a 16-bit A/B fragment from a row of f32 memory (global or LDS).
// ISA 16-bit fragment layout: element e of lane L holds
//   k(e) = (e < 8 ? e : e + 8) + 8 * (L >= 16)     (within the 32-wide K chunk)
// `rowk0` must be 16B aligned; loads lower to 4x (global|ds)_load_b128.
__device__ __forceinline__ v16h frag_from_row(const float* rowk0, int hi) {
  const float* p = rowk0 + hi * 8;
  float4 f0 = *(const float4*)(p);
  float4 f1 = *(const float4*)(p + 4);
  float4 f2 = *(const float4*)(p + 16);
  float4 f3 = *(const float4*)(p + 20);
  v16h r;
  r[0]  = (_Float16)f0.x; r[1]  = (_Float16)f0.y;
  r[2]  = (_Float16)f0.z; r[3]  = (_Float16)f0.w;
  r[4]  = (_Float16)f1.x; r[5]  = (_Float16)f1.y;
  r[6]  = (_Float16)f1.z; r[7]  = (_Float16)f1.w;
  r[8]  = (_Float16)f2.x; r[9]  = (_Float16)f2.y;
  r[10] = (_Float16)f2.z; r[11] = (_Float16)f2.w;
  r[12] = (_Float16)f3.x; r[13] = (_Float16)f3.y;
  r[14] = (_Float16)f3.z; r[15] = (_Float16)f3.w;
  return r;
}

// ---------------------------------------------------------------------------
// Kernel 1: fused QKV projection (+RoPE on q,k).
// GEMM: (128 x 4096) @ W^T, N = 4096(q)+1024(k)+1024(v) = 6144.
// One wave per 16x32 output tile (2 WMMAs / K-step, A reused); 1536 waves.
// q_ws layout (B,H,T,Dh); k_ws/v_ws layout (B,G,T,Dh).
// ---------------------------------------------------------------------------
__global__ __launch_bounds__(128) void qkv_rope_kernel(
    const float* __restrict__ x, const float* __restrict__ wq,
    const float* __restrict__ wk, const float* __restrict__ wv,
    const int* __restrict__ sp_ptr,
    float* __restrict__ q_ws, float* __restrict__ k_ws,
    float* __restrict__ v_ws) {
  const int wave  = blockIdx.x * 4 + (threadIdx.x >> 5);
  const int lane  = threadIdx.x & 31;
  const int hi    = lane >> 4;
  const int l15   = lane & 15;
  const int mtile = wave & 7;        // batch b (rows are b*16 + t)
  const int ntile = wave >> 3;       // 0..191
  const int n0    = ntile * 32;
  const int sp    = *sp_ptr;

  const float* wbase;
  int nrel;
  if (n0 < 4096)      { wbase = wq; nrel = n0; }
  else if (n0 < 5120) { wbase = wk; nrel = n0 - 4096; }
  else                { wbase = wv; nrel = n0 - 5120; }

  const float* arow  = x + (size_t)(mtile * 16 + l15) * 4096;
  const float* brow0 = wbase + (size_t)(nrel + l15) * 4096;
  const float* brow1 = brow0 + (size_t)16 * 4096;

  v8f c[2];
  c[0] = (v8f){}; c[1] = (v8f){};
  for (int k0 = 0; k0 < 4096; k0 += 32) {
    v16h a = frag_from_row(arow + k0, hi);
    c[0] = wmma_f16(a, frag_from_row(brow0 + k0, hi), c[0]);
    c[1] = wmma_f16(a, frag_from_row(brow1 + k0, hi), c[1]);
  }

  const int b = mtile;
#pragma unroll
  for (int nt = 0; nt < 2; nt++) {
    const int col  = n0 + nt * 16 + l15;   // C layout: lane holds one column
    const int d    = col & 127;
    const bool isv = (col >= 5120);
    const bool isk = (!isv) && (col >= 4096);
    if (isv) {
      const int g = (col - 5120) >> 7;
#pragma unroll
      for (int v = 0; v < 8; v++) {
        const int t = v + hi * 8;
        v_ws[((size_t)(b * 8 + g) * 16 + t) * 128 + d] = c[nt][v];
      }
    } else {
      // RoPE: pair (2i,2i+1); partner column sits in the adjacent lane.
      const int   i     = d >> 1;
      const bool  odd   = (d & 1);
      // inv_freq = 10000^(-i/64) = exp2(-i * log2(10000)/64)
      const float inv_f = exp2f(-(float)i * 0.20762050593046f);
#pragma unroll
      for (int v = 0; v < 8; v++) {
        const int   t   = v + hi * 8;
        const float ang = (float)(sp + t) * inv_f;
        float sn, cs;
        sincosf(ang, &sn, &cs);
        const float val   = c[nt][v];
        const float other = __shfl_xor(val, 1, 32);
        const float out   = odd ? (other * sn + val * cs)
                                : (val * cs - other * sn);
        if (isk) {
          const int g = (col - 4096) >> 7;
          k_ws[((size_t)(b * 8 + g) * 16 + t) * 128 + d] = out;
        } else {
          const int h = col >> 7;
          q_ws[((size_t)(b * 32 + h) * 16 + t) * 128 + d] = out;
        }
      }
    }
  }
}

// ---------------------------------------------------------------------------
// Kernel 2: flash attention over S with online softmax, split-K over S.
// grid = B*G*NSPLIT blocks, block = 128 threads (4 waves = 4 R-heads).
// Each 32-position K/V chunk is staged ONCE per block into LDS with
// GLOBAL_LOAD_ASYNC_TO_LDS_B128 and consumed by all 4 waves.
// Partials (O 16x128, rowmax, rowsum) per (b,g,r,split) go to workspace.
// ---------------------------------------------------------------------------
__global__ __launch_bounds__(128) void attn_kernel(
    const float* __restrict__ kc, const float* __restrict__ vc,
    const float* __restrict__ q_ws, const float* __restrict__ k_ws,
    const float* __restrict__ v_ws, const int* __restrict__ sp_ptr,
    float* __restrict__ O_part, float* __restrict__ rm_part,
    float* __restrict__ rs_part) {
  __shared__ float ldsK[32 * KV_PITCH];   // K chunk: 32 pos x 128 dh (padded)
  __shared__ float ldsV[32 * KV_PITCH];   // V chunk
  __shared__ float ldsP[4][16 * 33];      // per-wave P-transpose staging

  const int tid   = threadIdx.x;
  const int lane  = tid & 31;
  const int wv_   = tid >> 5;              // wave id == r (0..3)
  const int hi    = lane >> 4;
  const int l15   = lane & 15;
  const int bg    = blockIdx.x / NSPLIT;   // b*8+g
  const int split = blockIdx.x % NSPLIT;
  const int b     = bg >> 3;
  const int g     = bg & 7;
  const int sp    = *sp_ptr;
  const int r     = wv_;
  const int hq    = g * 4 + r;

  // Q A-fragments: 16 tokens x 128 dh (4 chunks of 32), loaded once.
  const float* qbase = q_ws + ((size_t)(b * 32 + hq) * 16) * 128;
  v16h qa[4];
#pragma unroll
  for (int ck = 0; ck < 4; ck++)
    qa[ck] = frag_from_row(qbase + (size_t)l15 * 128 + ck * 32, hi);

  const float* kcb = kc + (size_t)bg * 8192 * 128;
  const float* vcb = vc + (size_t)bg * 8192 * 128;
  const float* knb = k_ws + (size_t)bg * 16 * 128;
  const float* vnb = v_ws + (size_t)bg * 16 * 128;

  v8f   o[8];
  float rm[8], rs[8];
#pragma unroll
  for (int dt = 0; dt < 8; dt++) o[dt] = (v8f){};
#pragma unroll
  for (int v = 0; v < 8; v++) { rm[v] = -__builtin_inff(); rs[v] = 0.0f; }

  float* myp = &ldsP[wv_][0];
  const float scale = 0.08838834764831845f;  // 1/sqrt(128)
  const int s_begin = split * SCHUNK;

  for (int s0 = s_begin; s0 < s_begin + SCHUNK; s0 += 32) {
    // ---- cooperative async stage of K/V rows [s0, s0+32) into LDS ----
#pragma unroll
    for (int j = 0; j < 8; j++) {
      const int flat = tid + 128 * j;      // 0..1023 float4 units (32x32)
      const int srow = flat >> 5;
      const int c4   = flat & 31;
      const int s    = s0 + srow;
      const float* kr = (s < sp) ? (kcb + (size_t)s * 128)
                                 : (knb + (size_t)(s - sp) * 128);
      const float* vr = (s < sp) ? (vcb + (size_t)s * 128)
                                 : (vnb + (size_t)(s - sp) * 128);
      async_cp16(kr + c4 * 4, &ldsK[srow * KV_PITCH + c4 * 4]);
      async_cp16(vr + c4 * 4, &ldsV[srow * KV_PITCH + c4 * 4]);
    }
    wait_async0();
    __syncthreads();

    // ---- scores: two 16x16 tiles (K chunk of 32 positions), K from LDS ----
    v8f sc[2];
#pragma unroll
    for (int tI = 0; tI < 2; tI++) {
      const int srow = tI * 16 + l15;      // this lane's key position (local)
      const int s    = s0 + srow;
      const float* krow = &ldsK[srow * KV_PITCH];
      v8f acc = {};
#pragma unroll
      for (int ck = 0; ck < 4; ck++) {
        v16h kb = frag_from_row(krow + ck * 32, hi);
        acc = wmma_f16(qa[ck], kb, acc);
      }
#pragma unroll
      for (int v = 0; v < 8; v++) {
        float val = acc[v] * scale;
        const int t = v + hi * 8;          // query token of this row
        if (s > sp + t) val = -__builtin_inff();  // causal mask
        acc[v] = val;
      }
      sc[tI] = acc;
    }

    // ---- online softmax update (per row; rows live per-VGPR per half) ----
#pragma unroll
    for (int v = 0; v < 8; v++) {
      float m = fmaxf(sc[0][v], sc[1][v]);
#pragma unroll
      for (int msk = 1; msk < 16; msk <<= 1)
        m = fmaxf(m, __shfl_xor(m, msk, 32));
      float nm = fmaxf(rm[v], m);
      nm = fmaxf(nm, -1e30f);              // keep exp(-inf - nm) well-defined
      const float corr = __expf(rm[v] - nm);
      const float p0 = __expf(sc[0][v] - nm);
      const float p1 = __expf(sc[1][v] - nm);
      float psum = p0 + p1;
#pragma unroll
      for (int msk = 1; msk < 16; msk <<= 1)
        psum += __shfl_xor(psum, msk, 32);
      rs[v] = rs[v] * corr + psum;
      rm[v] = nm;
#pragma unroll
      for (int dt = 0; dt < 8; dt++) o[dt][v] *= corr;
      // stage P (C layout -> LDS; row = token, col = s offset 0..31)
      myp[(v + hi * 8) * 33 + l15]      = p0;
      myp[(v + hi * 8) * 33 + 16 + l15] = p1;
    }
    __syncthreads();

    // ---- P A-fragment from LDS; V B-fragments from LDS; O += P @ V ----
    v16h pf;
    {
      const float* row = myp + l15 * 33;   // lane row = token t
#pragma unroll
      for (int e = 0; e < 16; e++) {
        const int k = (e < 8 ? e : e + 8) + hi * 8;
        pf[e] = (_Float16)row[k];
      }
    }
#pragma unroll
    for (int dt = 0; dt < 8; dt++) {
      v16h vf;
#pragma unroll
      for (int e = 0; e < 16; e++) {
        const int srel = (e < 8 ? e : e + 8) + hi * 8;
        vf[e] = (_Float16)ldsV[srel * KV_PITCH + dt * 16 + l15];
      }
      o[dt] = wmma_f16(pf, vf, o[dt]);
    }
    __syncthreads();   // protect ldsK/V/P before next iteration's writes
  }

  // ---- store partials ----
  const int p = (bg * 4 + r) * NSPLIT + split;
  float* ob = O_part + (size_t)p * 2048;
#pragma unroll
  for (int dt = 0; dt < 8; dt++)
#pragma unroll
    for (int v = 0; v < 8; v++)
      ob[(size_t)(v + hi * 8) * 128 + dt * 16 + l15] = o[dt][v];
  if (l15 == 0) {
#pragma unroll
    for (int v = 0; v < 8; v++) {
      rm_part[(size_t)p * 16 + v + hi * 8] = rm[v];
      rs_part[(size_t)p * 16 + v + hi * 8] = rs[v];
    }
  }
}

// ---------------------------------------------------------------------------
// Kernel 3: combine the NSPLIT partial softmax results; write attn activations
// in (b, t, h*128+dh) layout (h = g*4 + r) for the output GEMM.
// ---------------------------------------------------------------------------
__global__ __launch_bounds__(128) void combine_kernel(
    const float* __restrict__ O_part, const float* __restrict__ rm_part,
    const float* __restrict__ rs_part, float* __restrict__ attn) {
  __shared__ float coef[NSPLIT][16];
  const int bgr = blockIdx.x;          // ((b*8+g)*4 + r)
  const int tid = threadIdx.x;

  if (tid < 16) {
    float ms[NSPLIT];
    float gm = -__builtin_inff();
#pragma unroll
    for (int i = 0; i < NSPLIT; i++) {
      ms[i] = rm_part[(size_t)(bgr * NSPLIT + i) * 16 + tid];
      gm = fmaxf(gm, ms[i]);
    }
    float denom = 0.0f;
    float ex[NSPLIT];
#pragma unroll
    for (int i = 0; i < NSPLIT; i++) {
      ex[i] = __expf(ms[i] - gm);
      denom += rs_part[(size_t)(bgr * NSPLIT + i) * 16 + tid] * ex[i];
    }
    const float inv = 1.0f / denom;
#pragma unroll
    for (int i = 0; i < NSPLIT; i++) coef[i][tid] = ex[i] * inv;
  }
  __syncthreads();

  const int b = bgr >> 5;
  const int g = (bgr >> 2) & 7;
  const int r = bgr & 3;
  for (int idx = tid; idx < 2048; idx += 128) {
    const int t = idx >> 7, dh = idx & 127;
    float acc = 0.0f;
#pragma unroll
    for (int i = 0; i < NSPLIT; i++)
      acc += O_part[(size_t)(bgr * NSPLIT + i) * 2048 + idx] * coef[i][t];
    attn[((size_t)(b * 16 + t)) * 4096 + (g * 4 + r) * 128 + dh] = acc;
  }
}

// ---------------------------------------------------------------------------
// Kernel 4: output projection (128 x 4096) @ wo^T -> d_out.
// One wave per 16x32 tile (2 WMMAs / K-step, A reused); 1024 waves.
// ---------------------------------------------------------------------------
__global__ __launch_bounds__(128) void out_proj_kernel(
    const float* __restrict__ attn, const float* __restrict__ wo,
    float* __restrict__ out) {
  const int wave  = blockIdx.x * 4 + (threadIdx.x >> 5);
  const int lane  = threadIdx.x & 31;
  const int hi    = lane >> 4;
  const int l15   = lane & 15;
  const int mtile = wave & 7;
  const int ntile = wave >> 3;   // 0..127
  const int n0    = ntile * 32;

  const float* arow  = attn + (size_t)(mtile * 16 + l15) * 4096;
  const float* brow0 = wo + (size_t)(n0 + l15) * 4096;
  const float* brow1 = brow0 + (size_t)16 * 4096;

  v8f c[2];
  c[0] = (v8f){}; c[1] = (v8f){};
  for (int k0 = 0; k0 < 4096; k0 += 32) {
    v16h a = frag_from_row(arow + k0, hi);
    c[0] = wmma_f16(a, frag_from_row(brow0 + k0, hi), c[0]);
    c[1] = wmma_f16(a, frag_from_row(brow1 + k0, hi), c[1]);
  }
#pragma unroll
  for (int nt = 0; nt < 2; nt++)
#pragma unroll
    for (int v = 0; v < 8; v++)
      out[(size_t)(mtile * 16 + v + hi * 8) * 4096 + n0 + nt * 16 + l15] =
          c[nt][v];
}

// ---------------------------------------------------------------------------
// Host launch. Inputs (per reference setup_inputs order):
//  0:x  1:key_cache  2:value_cache  3:wq  4:wk  5:wv  6:wo  7:start_pos(int)
// Workspace (floats): q_ws 524288 | k_ws 131072 | v_ws 131072 |
//   attn 524288 | O_part 4194304 | rm 32768 | rs 32768  (~22.3 MB total)
// ---------------------------------------------------------------------------
extern "C" void kernel_launch(void* const* d_in, const int* in_sizes, int n_in,
                              void* d_out, int out_size, void* d_ws,
                              size_t ws_size, hipStream_t stream) {
  const float* x  = (const float*)d_in[0];
  const float* kc = (const float*)d_in[1];
  const float* vc = (const float*)d_in[2];
  const float* wq = (const float*)d_in[3];
  const float* wk = (const float*)d_in[4];
  const float* wv = (const float*)d_in[5];
  const float* wo = (const float*)d_in[6];
  const int*   sp = (const int*)d_in[7];
  float* out = (float*)d_out;

  float* w = (float*)d_ws;
  float* q_ws    = w;                 // 524288
  float* k_ws    = w + 524288;        // 131072
  float* v_ws    = w + 655360;        // 131072
  float* attn_ws = w + 786432;        // 524288
  float* O_part  = w + 1310720;       // 4194304
  float* rm_part = w + 5505024;       // 32768
  float* rs_part = w + 5537792;       // 32768

  qkv_rope_kernel<<<384, 128, 0, stream>>>(x, wq, wk, wv, sp, q_ws, k_ws, v_ws);
  attn_kernel<<<8 * 8 * NSPLIT, 128, 0, stream>>>(kc, vc, q_ws, k_ws, v_ws, sp,
                                                  O_part, rm_part, rs_part);
  combine_kernel<<<256, 128, 0, stream>>>(O_part, rm_part, rs_part, attn_ws);
  out_proj_kernel<<<256, 128, 0, stream>>>(attn_ws, wo, out);
}
